// DeformableCrossAttention_61074434949222
// MI455X (gfx1250) — matrix operand order repaired and checked by reading, not compiled, not run on hardware
//
#include <hip/hip_runtime.h>
#include <hip/hip_bf16.h>
#include <math.h>

// ---------------------------------------------------------------------------
// Deformable cross-attention, algebraically refactored for MI455X (gfx1250).
//
//   scores[h,k] = sampled[k] . qt[h]      with qt = q_h @ Wk_h^T   (hoisted)
//   out_h       = (sum_k attn[h,k] sampled[k]) @ Wv_h + bv_h       (hoisted)
//
// GEMMs: v_wmma_f32_16x16x32_bf16, branch-free b128-vectorized fragment
// loads, one wave owns a 16x(16*NT) strip so the A fragment feeds NT WMMAs.
// Ragged N=98 GEMM runs against a zero-padded N=112 weight copy (no guards).
// ---------------------------------------------------------------------------

#define Bsz 4
#define Nq  1024
#define Dm  256
#define NHd 8
#define HD  32
#define KK  49
#define HH_ 64
#define WW_ 64
#define OFFP 112   // padded 2*KK (=98 -> 112, multiple of 16)

typedef __attribute__((ext_vector_type(16))) __bf16 v16bf;
typedef __attribute__((ext_vector_type(8)))  float  v8f;
typedef __attribute__((ext_vector_type(4)))  float  v4f;

// ---------------------------------------------------------------------------
// Batched WMMA GEMM:  C[z] = act( A[z] (MxK) * B[z] (KxN) + bias[z] )
//  One wave computes a 16x(16*NT) strip: A fragment loaded once per k-step,
//  reused for NT WMMAs.
//  BCONTIG=1: B's k-index is memory-contiguous (transposed weight view),
//             ldb = column stride  -> 4x b128 loads per B fragment.
//  BCONTIG=0: B row-major, ldb = row stride -> 16 lane-coalesced b32 loads.
//  ACT: 0 none, 1 relu, 2 tanh(x)*(*scale_ptr)   (compile-time)
//  NO bounds checks: M%16==0, N%(16*NT)==0, K%32==0 required.
// ---------------------------------------------------------------------------
template<int BCONTIG, int NT, int ACT>
__global__ __launch_bounds__(256) void gemm_bf16_wmma(
    const float* __restrict__ A, const float* __restrict__ Bm,
    const float* __restrict__ bias, float* __restrict__ C,
    const float* __restrict__ scale_ptr,
    int M, int N, int K,
    long a_off0, long lda, long a_bs,
    long b_off0, long ldb, long b_bs,
    long c_off0, long ldc, long c_bs,
    long bias_off0, long bias_bs)
{
  const int lane   = threadIdx.x & 31;
  const int widb   = threadIdx.x >> 5;
  const int groups = N / (16 * NT);
  const int tiles_m = M >> 4;
  const int wid = blockIdx.x * (blockDim.x >> 5) + widb;
  if (wid >= tiles_m * groups) return;            // whole-wave uniform exit
  const int tg = wid % groups;
  const int tm = wid / groups;
  const long z = blockIdx.y;

  const float* Ab = A  + a_off0 + z * a_bs;
  const float* Bb = Bm + b_off0 + z * b_bs;
  float*       Cb = C  + c_off0 + z * c_bs;
  const float* biasb = bias ? (bias + bias_off0 + z * bias_bs) : nullptr;

  const int half = lane >> 4;        // which 16-lane half of the wave
  const int l15  = lane & 15;
  const int am   = tm * 16 + l15;    // A row this lane feeds
  const int bn0  = tg * NT * 16 + l15;

  // Per-lane base pointers (k = 16*half + s, same order for A and B).
  const float* Arow = Ab + (long)am * lda + half * 16;  // one contiguous 16-run
  const float* Bbase[NT];
#pragma unroll
  for (int t = 0; t < NT; ++t) {
    const int bn = bn0 + t * 16;
    Bbase[t] = BCONTIG ? (Bb + (long)bn * ldb + half * 16)
                       : (Bb + bn + (long)(half * 16) * ldb);
  }

  v8f acc[NT];
#pragma unroll
  for (int t = 0; t < NT; ++t) { v8f zz = {}; acc[t] = zz; }

  for (int k0 = 0; k0 < K; k0 += 32) {
    // ---- A fragment: one contiguous 16-float run -> 4x b128, shared -----
    v4f a0 = *(const v4f*)(Arow + k0);
    v4f a1 = *(const v4f*)(Arow + k0 + 4);
    v4f a2 = *(const v4f*)(Arow + k0 + 8);
    v4f a3 = *(const v4f*)(Arow + k0 + 12);
    v16bf afrag;
#pragma unroll
    for (int e = 0; e < 4; ++e) {
      afrag[e]      = (__bf16)a0[e];
      afrag[4 + e]  = (__bf16)a1[e];
      afrag[8 + e]  = (__bf16)a2[e];
      afrag[12 + e] = (__bf16)a3[e];
    }
    // ---- NT B fragments --------------------------------------------------
    v16bf bfrag[NT];
#pragma unroll
    for (int t = 0; t < NT; ++t) {
      if (BCONTIG) {
        v4f b0 = *(const v4f*)(Bbase[t] + k0);
        v4f b1 = *(const v4f*)(Bbase[t] + k0 + 4);
        v4f b2 = *(const v4f*)(Bbase[t] + k0 + 8);
        v4f b3 = *(const v4f*)(Bbase[t] + k0 + 12);
#pragma unroll
        for (int e = 0; e < 4; ++e) {
          bfrag[t][e]      = (__bf16)b0[e];
          bfrag[t][4 + e]  = (__bf16)b1[e];
          bfrag[t][8 + e]  = (__bf16)b2[e];
          bfrag[t][12 + e] = (__bf16)b3[e];
        }
      } else {
        const float* Bp = Bbase[t] + (long)k0 * ldb;
#pragma unroll
        for (int e = 0; e < 16; ++e)
          bfrag[t][e] = (__bf16)Bp[(long)e * ldb];
      }
    }
    // ---- NT WMMAs reuse the one A fragment ------------------------------
#pragma unroll
    for (int t = 0; t < NT; ++t)
      acc[t] = __builtin_amdgcn_wmma_f32_16x16x32_bf16(
          /*neg_a=*/false, afrag, /*neg_b=*/false, bfrag[t],
          /*c_mod=*/(short)0, acc[t], /*reuse_a=*/false, /*reuse_b=*/false);
  }

  const float scale = (ACT == 2) ? scale_ptr[0] : 1.0f;
#pragma unroll
  for (int t = 0; t < NT; ++t) {
    const int bn = bn0 + t * 16;
    const float bb = biasb ? biasb[bn] : 0.0f;
#pragma unroll
    for (int r = 0; r < 8; ++r) {
      int m = tm * 16 + r + half * 8;   // C/D ISA layout
      float v = acc[t][r] + bb;
      if (ACT == 1) v = fmaxf(v, 0.0f);
      else if (ACT == 2) v = tanhf(v) * scale;
      Cb[(long)m * ldc + bn] = v;
    }
  }
}

// ---------------------------------------------------------------------------
// Zero-pad Woff2 (256x98 -> 256x112) and boff2 (98 -> 112) into workspace.
// ---------------------------------------------------------------------------
__global__ void pad_off_weights(const float* __restrict__ Woff2,
                                const float* __restrict__ boff2,
                                float* __restrict__ Wp, float* __restrict__ bp)
{
  int idx = blockIdx.x * blockDim.x + threadIdx.x;
  const int total = Dm * OFFP;
  if (idx < total) {
    int row = idx / OFFP, col = idx % OFFP;
    Wp[idx] = (col < 2 * KK) ? Woff2[row * (2 * KK) + col] : 0.0f;
  }
  if (idx < OFFP) bp[idx] = (idx < 2 * KK) ? boff2[idx] : 0.0f;
}

// ---------------------------------------------------------------------------
// Fused: bilinear sampling -> scores (vs hoisted qt) -> softmax -> per-head
// weighted aggregation.  One block per query token.  LDS ~59 KB.
// ---------------------------------------------------------------------------
__global__ __launch_bounds__(256) void deform_sample_attend(
    const float* __restrict__ ref_boxes,  // (B,N,4)
    const float* __restrict__ mem,        // (B,C,H,W)
    const float* __restrict__ off,        // (B*N, OFFP)  [k*2+{x,y}]
    const float* __restrict__ qt,         // (B*N, NH, C)
    const float* __restrict__ qvec,       // (B*N, D)
    const float* __restrict__ bk,         // (D)
    float* __restrict__ agg,              // (B*N, NH, C)
    float* __restrict__ attn_out)         // (B, NH, N, K)
{
  __shared__ float s_samp[KK * Dm];   // 49 x 256 sampled features
  __shared__ float s_qt[NHd * Dm];    // hoisted query-key rows
  __shared__ float s_attn[NHd * KK];  // scores -> attn in place
  __shared__ float s_qb[NHd];         // q . bk bias term per head

  const int tid = threadIdx.x;
  const long bn = blockIdx.x;               // 0 .. B*N-1
  const int  b  = (int)(bn / Nq);
  const int  nq = (int)(bn % Nq);

  // ---- stage qt row into LDS -------------------------------------------
  for (int i = tid; i < NHd * Dm; i += 256)
    s_qt[i] = qt[bn * (NHd * Dm) + i];

  // ---- q . bk per head --------------------------------------------------
  if (tid < NHd) {
    float s = 0.0f;
    const float* qrow = qvec + bn * Dm + tid * HD;
    const float* bkh  = bk + tid * HD;
    for (int d = 0; d < HD; ++d) s += qrow[d] * bkh[d];
    s_qb[tid] = s;
  }

  // ---- bilinear sampling: thread = channel, loop over 49 points ---------
  const float cx = ref_boxes[bn * 4 + 0] * 2.0f - 1.0f;
  const float cy = ref_boxes[bn * 4 + 1] * 2.0f - 1.0f;
  const int c = tid;                      // 256 channels == 256 threads
  const float* mch = mem + ((long)(b * Dm + c)) * (HH_ * WW_);
  for (int k = 0; k < KK; ++k) {
    float lx = (float)((k % 7) - 3) * (2.0f / 63.0f);
    float ly = (float)((k / 7) - 3) * (2.0f / 63.0f);
    float sx = cx + lx + off[bn * OFFP + k * 2 + 0];
    float sy = cy + ly + off[bn * OFFP + k * 2 + 1];
    sx = fminf(fmaxf(sx, -1.0f), 1.0f);
    sy = fminf(fmaxf(sy, -1.0f), 1.0f);
    float x = (sx + 1.0f) * 0.5f * (WW_ - 1);
    float y = (sy + 1.0f) * 0.5f * (HH_ - 1);
    float x0 = floorf(x), y0 = floorf(y);
    float fx = x - x0, fy = y - y0;
    int x0i = min(max((int)x0, 0), WW_ - 1);
    int x1i = min(x0i + 1, WW_ - 1);
    int y0i = min(max((int)y0, 0), HH_ - 1);
    int y1i = min(y0i + 1, HH_ - 1);
    float m00 = mch[y0i * WW_ + x0i];
    float m01 = mch[y0i * WW_ + x1i];
    float m10 = mch[y1i * WW_ + x0i];
    float m11 = mch[y1i * WW_ + x1i];
    s_samp[k * Dm + c] = m00 * (1.0f - fx) * (1.0f - fy)
                       + m01 * fx * (1.0f - fy)
                       + m10 * (1.0f - fx) * fy
                       + m11 * fx * fy;
  }
  __syncthreads();

  // ---- scores[h,k] = (sampled[k] . qt[h] + q.bk[h]) / sqrt(hd) ----------
  const float inv_sqrt_hd = 0.17677669529663687f;  // 1/sqrt(32)
  for (int t = tid; t < NHd * KK; t += 256) {
    int h = t / KK, k = t % KK;
    const float* sp = s_samp + k * Dm;
    const float* qp = s_qt + h * Dm;
    float dot = 0.0f;
    for (int cc = 0; cc < Dm; ++cc) dot += sp[cc] * qp[cc];
    s_attn[t] = (dot + s_qb[h]) * inv_sqrt_hd;
  }
  __syncthreads();

  // ---- softmax over k per head, write attn tuple output -----------------
  if (tid < NHd) {
    const int h = tid;
    float mx = -1e30f;
    for (int k = 0; k < KK; ++k) mx = fmaxf(mx, s_attn[h * KK + k]);
    float sum = 0.0f;
    for (int k = 0; k < KK; ++k) {
      float e = __expf(s_attn[h * KK + k] - mx);
      s_attn[h * KK + k] = e;
      sum += e;
    }
    float inv = 1.0f / sum;
    float* ao = attn_out + (((long)b * NHd + h) * Nq + nq) * KK;
    for (int k = 0; k < KK; ++k) {
      float a = s_attn[h * KK + k] * inv;
      s_attn[h * KK + k] = a;
      ao[k] = a;
    }
  }
  __syncthreads();

  // ---- agg[h,c] = sum_k attn[h,k] * sampled[k,c] -------------------------
  for (int t = tid; t < NHd * Dm; t += 256) {
    int h = t >> 8, cc = t & 255;
    float s = 0.0f;
    for (int k = 0; k < KK; ++k) s += s_attn[h * KK + k] * s_samp[k * Dm + cc];
    agg[bn * (NHd * Dm) + t] = s;
  }
}

// ---------------------------------------------------------------------------
// Host-side launch
// ---------------------------------------------------------------------------
extern "C" void kernel_launch(void* const* d_in, const int* in_sizes, int n_in,
                              void* d_out, int out_size, void* d_ws, size_t ws_size,
                              hipStream_t stream) {
  const float* X      = (const float*)d_in[0];   // decoder_embed (B,N,D)
  const float* boxes  = (const float*)d_in[1];   // ref_boxes (B,N,4)
  const float* mem    = (const float*)d_in[2];   // memory (B,D,H,W)
  const float* Wq     = (const float*)d_in[3];
  const float* bq     = (const float*)d_in[4];
  const float* Wk     = (const float*)d_in[5];
  const float* bk     = (const float*)d_in[6];
  const float* Wv     = (const float*)d_in[7];
  const float* bv     = (const float*)d_in[8];
  const float* Wo     = (const float*)d_in[9];
  const float* bo     = (const float*)d_in[10];
  const float* Woff1  = (const float*)d_in[11];
  const float* boff1  = (const float*)d_in[12];
  const float* Woff2  = (const float*)d_in[13];
  const float* boff2  = (const float*)d_in[14];
  const float* sigma  = (const float*)d_in[15];

  float* out  = (float*)d_out;                         // (B,N,D)
  float* attn = out + (long)Bsz * Nq * Dm;             // (B,NH,N,K)

  const long BN = (long)Bsz * Nq;
  float* ws      = (float*)d_ws;
  float* ws_q    = ws;                       // BN*D
  float* ws_h    = ws_q   + BN * Dm;         // BN*D
  float* ws_off  = ws_h   + BN * Dm;         // BN*OFFP
  float* ws_qt   = ws_off + BN * OFFP;       // BN*NH*D
  float* ws_agg  = ws_qt  + BN * NHd * Dm;   // BN*NH*D
  float* ws_oh   = ws_agg + BN * NHd * Dm;   // BN*D
  float* ws_w2p  = ws_oh  + BN * Dm;         // Dm*OFFP (padded Woff2)
  float* ws_b2p  = ws_w2p + (long)Dm * OFFP; // OFFP    (padded boff2)

  const dim3 blk(256);
  auto blocks_for = [](int M, int N, int NT) {
    int tiles = (M / 16) * (N / (16 * NT));
    return (tiles + 7) / 8;                 // 8 waves / block
  };

  // 0) zero-pad offset-head weights to N=112
  pad_off_weights<<<dim3((Dm * OFFP + 255) / 256), blk, 0, stream>>>(
      Woff2, boff2, ws_w2p, ws_b2p);

  // 1) q = X @ Wq + bq                      (4096x256x256, NT=4)
  gemm_bf16_wmma<0, 4, 0><<<dim3(blocks_for(4096, 256, 4), 1), blk, 0, stream>>>(
      X, Wq, bq, ws_q, nullptr, 4096, 256, 256,
      0, 256, 0,   0, 256, 0,   0, 256, 0,   0, 0);

  // 2) h = relu(X @ Woff1 + boff1)          (NT=4, relu epilogue)
  gemm_bf16_wmma<0, 4, 1><<<dim3(blocks_for(4096, 256, 4), 1), blk, 0, stream>>>(
      X, Woff1, boff1, ws_h, nullptr, 4096, 256, 256,
      0, 256, 0,   0, 256, 0,   0, 256, 0,   0, 0);

  // 3) off = tanh(h @ Woff2p + boff2p) * sigma   (padded N=112, NT=1)
  gemm_bf16_wmma<0, 1, 2><<<dim3(blocks_for(4096, OFFP, 1), 1), blk, 0, stream>>>(
      ws_h, ws_w2p, ws_b2p, ws_off, sigma, 4096, OFFP, 256,
      0, 256, 0,   0, OFFP, 0,   0, OFFP, 0,   0, 0);

  // 4) qt[h] = q_h @ Wk_h^T   (z = head; B k-contiguous -> b128, NT=4)
  gemm_bf16_wmma<1, 4, 0><<<dim3(blocks_for(4096, 256, 4), NHd), blk, 0, stream>>>(
      ws_q, Wk, nullptr, ws_qt, nullptr, 4096, 256, 32,
      0, 256, 32,              // A: q, head offset h*32
      0, 256, 32,              // B: Wk[n*256 + h*32 + k], ldb = col stride 256
      0, (long)NHd * Dm, 256,  // C: qt row stride 2048, head offset h*256
      0, 0);

  // 5) fused sampling + scores + softmax + aggregation
  deform_sample_attend<<<dim3((unsigned)BN), blk, 0, stream>>>(
      boxes, mem, ws_off, ws_qt, ws_q, bk, ws_agg, attn);

  // 6) out_heads[h] = agg_h @ Wv_h + bv_h    (z = head, N = 32, NT=2)
  gemm_bf16_wmma<0, 2, 0><<<dim3(blocks_for(4096, 32, 2), NHd), blk, 0, stream>>>(
      ws_agg, Wv, bv, ws_oh, nullptr, 4096, 32, 256,
      0, (long)NHd * Dm, 256,  // A: agg, head offset h*256, row stride 2048
      0, 256, 32,              // B: Wv rows, columns h*32..h*32+31
      0, 256, 32,              // C: out_heads columns h*32..h*32+31
      0, 32);

  // 7) out = out_heads @ Wo + bo  -> d_out  (NT=4)
  gemm_bf16_wmma<0, 4, 0><<<dim3(blocks_for(4096, 256, 4), 1), blk, 0, stream>>>(
      ws_oh, Wo, bo, out, nullptr, 4096, 256, 256,
      0, 256, 0,   0, 256, 0,   0, 256, 0,   0, 0);
}
